// CrossAttention_72112500900069
// MI455X (gfx1250) — matrix operand order repaired and checked by reading, not compiled, not run on hardware
//
#include <hip/hip_runtime.h>
#include <hip/hip_bf16.h>
#include <stdint.h>

// ---------------- constants ----------------
#define B_    2
#define T_    2048
#define C_    1024
#define H_    16
#define HD_   64
#define HID_  1365
#define HIDP_ 1376          // HID padded to multiple of 32 for WMMA K
#define BT_   (B_*T_)
#define EPS_  1e-5f
#define L2E_  1.44269504088896340736f

typedef __attribute__((ext_vector_type(16))) __bf16 v16bf;
typedef __attribute__((ext_vector_type(8)))  float  v8f;

union Frag { v16bf v; uint4 u[2]; unsigned short s[16]; };

__device__ inline unsigned short f2bf(float x) {
  unsigned u = __float_as_uint(x);
  unsigned r = u + 0x7FFFu + ((u >> 16) & 1u);
  return (unsigned short)(r >> 16);
}

__device__ inline v8f vzero() {
  v8f z;
#pragma unroll
  for (int i = 0; i < 8; ++i) z[i] = 0.f;
  return z;
}

// A-matrix (16x32 bf16) per-lane layout: row = lane&15,
// half 0: K {0..7, 16..23}; half 1: K {8..15, 24..31}.
__device__ inline void loadA(Frag& f, const unsigned short* row, int k, int half) {
  int koff = half * 8;
  f.u[0] = *reinterpret_cast<const uint4*>(row + k + koff);
  f.u[1] = *reinterpret_cast<const uint4*>(row + k + koff + 16);
}
// B-matrix (32x16 bf16) per-lane layout: col = lane&15,
// half 0: K {0..7, 16..23}; half 1: K {8..15, 24..31}.
__device__ inline void loadB(Frag& f, const unsigned short* row, int k, int half) {
  const unsigned short* p = row + k + half * 8;
  f.u[0] = *reinterpret_cast<const uint4*>(p);
  f.u[1] = *reinterpret_cast<const uint4*>(p + 16);
}
__device__ inline v8f wmma_bf(const Frag& a, const Frag& b, v8f c) {
  return __builtin_amdgcn_wmma_f32_16x16x32_bf16(false, a.v, false, b.v,
                                                 (short)0, c, false, false);
}

// ---------------- LayerNorm ----------------
template <int OUT_BF>
__global__ void k_ln(const float* __restrict__ x, const float* __restrict__ g,
                     const float* __restrict__ be, void* __restrict__ out) {
  __shared__ float s1[256], s2[256];
  int row = blockIdx.x, t = threadIdx.x;
  const float* xr = x + (size_t)row * C_;
  float a = 0.f, b = 0.f;
  for (int i = t; i < C_; i += 256) { float v = xr[i]; a += v; b += v * v; }
  s1[t] = a; s2[t] = b; __syncthreads();
  for (int o = 128; o > 0; o >>= 1) {
    if (t < o) { s1[t] += s1[t + o]; s2[t] += s2[t + o]; }
    __syncthreads();
  }
  float mean = s1[0] * (1.f / C_);
  float var  = s2[0] * (1.f / C_) - mean * mean;
  float inv  = rsqrtf(var + EPS_);
  for (int i = t; i < C_; i += 256) {
    float v = (xr[i] - mean) * inv * g[i] + be[i];
    if (OUT_BF) ((unsigned short*)out)[(size_t)row * C_ + i] = f2bf(v);
    else        ((float*)out)[(size_t)row * C_ + i] = v;
  }
}

// ---------------- fp32 -> bf16 converts ----------------
__global__ void k_conv(const float* __restrict__ src, unsigned short* __restrict__ dst, int n) {
  int i = blockIdx.x * 256 + threadIdx.x;
  if (i < n) dst[i] = f2bf(src[i]);
}
// w3 [C, HID] -> [C, HIDP] zero-padded K
__global__ void k_convpad(const float* __restrict__ src, unsigned short* __restrict__ dst) {
  int i = blockIdx.x * 256 + threadIdx.x;
  if (i >= C_ * HIDP_) return;
  int r = i / HIDP_, c = i - r * HIDP_;
  dst[i] = (c < HID_) ? f2bf(src[(size_t)r * HID_ + c]) : (unsigned short)0;
}
// mean over heads of rel_table -> meanrel[4095]
__global__ void k_meanrel(const float* __restrict__ rel, float* __restrict__ mr) {
  int j = blockIdx.x * 256 + threadIdx.x;
  if (j >= 2 * T_ - 1) return;
  float s = 0.f;
#pragma unroll
  for (int h = 0; h < H_; ++h) s += rel[(size_t)h * (2 * T_ - 1) + j];
  mr[j] = s * (1.f / H_);
}

// ---------------- GEMM core (unguarded): one wave -> 16(M) x 64(N) ----------------
// N tiles must be fully in-range. Straight-line k-loop: no exec manipulation.
__device__ inline void gemm_core4(const unsigned short* __restrict__ A, int lda,
                                  const unsigned short* __restrict__ W, int ldw,
                                  int mbase, int nbase, int K, v8f acc[4]) {
  int lane = threadIdx.x & 31, m = lane & 15, half = lane >> 4;
  const unsigned short* arow = A + (size_t)(mbase + m) * lda;
  const unsigned short* wr0 = W + (size_t)(nbase + m) * ldw;
  const unsigned short* wr1 = wr0 + (size_t)16 * ldw;
  const unsigned short* wr2 = wr0 + (size_t)32 * ldw;
  const unsigned short* wr3 = wr0 + (size_t)48 * ldw;
  for (int k = 0; k < K; k += 32) {
    Frag a; loadA(a, arow, k, half);
    Frag b0, b1, b2, b3;
    loadB(b0, wr0, k, half);
    loadB(b1, wr1, k, half);
    loadB(b2, wr2, k, half);
    loadB(b3, wr3, k, half);
    acc[0] = wmma_bf(a, b0, acc[0]);
    acc[1] = wmma_bf(a, b1, acc[1]);
    acc[2] = wmma_bf(a, b2, acc[2]);
    acc[3] = wmma_bf(a, b3, acc[3]);
  }
}

// QKV projection: A = qln (q section) / kln (k,v sections); scatter to per-head layouts.
__global__ void k_gemm_qkv(const unsigned short* __restrict__ qln,
                           const unsigned short* __restrict__ kln,
                           const unsigned short* __restrict__ W,
                           const float* __restrict__ bias,
                           unsigned short* __restrict__ qp,
                           unsigned short* __restrict__ kp,
                           unsigned short* __restrict__ vT) {
  const int NT = (3 * C_) / 64;                       // 48
  int widx = blockIdx.x * 4 + (threadIdx.x >> 5);
  int ntile = widx % NT, mtile = widx / NT;
  int nbase = ntile * 64, mbase = mtile * 16;
  int which = nbase / C_;                             // 0=q 1=k 2=v
  const unsigned short* A = (which == 0) ? qln : kln;
  v8f acc[4];
#pragma unroll
  for (int i = 0; i < 4; ++i) acc[i] = vzero();
  gemm_core4(A, C_, W, C_, mbase, nbase, C_, acc);

  int lane = threadIdx.x & 31, half = lane >> 4;
#pragma unroll
  for (int nt = 0; nt < 4; ++nt) {
    int n = nbase + nt * 16 + (lane & 15);
    int cs = n & (C_ - 1), h = cs >> 6, d = cs & 63;
#pragma unroll
    for (int r = 0; r < 8; ++r) {
      int row = mbase + r + half * 8;
      int b = row / T_, t = row - b * T_;
      float v = acc[nt][r] + bias[n];
      if (which == 0) { v *= 0.125f;  // 1/sqrt(HD)
        qp[(((size_t)(b * H_ + h)) * T_ + t) * HD_ + d] = f2bf(v);
      } else if (which == 1) {
        kp[(((size_t)(b * H_ + h)) * T_ + t) * HD_ + d] = f2bf(v);
      } else {
        vT[(((size_t)(b * H_ + h)) * HD_ + d) * T_ + t] = f2bf(v);
      }
    }
  }
}

// Generic GEMM -> bf16 out (+bias), N multiple of 64
__global__ void k_gemm_bf16out(const unsigned short* __restrict__ A, int lda,
                               const unsigned short* __restrict__ W, int ldw, int K,
                               const float* __restrict__ bias,
                               unsigned short* __restrict__ out, int ldo, int NT) {
  int widx = blockIdx.x * 4 + (threadIdx.x >> 5);
  int ntile = widx % NT, mtile = widx / NT;
  int nbase = ntile * 64, mbase = mtile * 16;
  v8f acc[4];
#pragma unroll
  for (int i = 0; i < 4; ++i) acc[i] = vzero();
  gemm_core4(A, lda, W, ldw, mbase, nbase, K, acc);
  int lane = threadIdx.x & 31, half = lane >> 4;
#pragma unroll
  for (int nt = 0; nt < 4; ++nt) {
    int col = nbase + nt * 16 + (lane & 15);
    float bb = bias ? bias[col] : 0.f;
#pragma unroll
    for (int r = 0; r < 8; ++r) {
      int row = mbase + r + half * 8;
      out[(size_t)row * ldo + col] = f2bf(acc[nt][r] + bb);
    }
  }
}

// Generic GEMM -> f32 out, optional bias + residual add, N multiple of 64
__global__ void k_gemm_f32res(const unsigned short* __restrict__ A, int lda,
                              const unsigned short* __restrict__ W, int ldw, int K,
                              const float* __restrict__ bias,
                              const float* __restrict__ res,
                              float* __restrict__ out, int ldo, int NT) {
  int widx = blockIdx.x * 4 + (threadIdx.x >> 5);
  int ntile = widx % NT, mtile = widx / NT;
  int nbase = ntile * 64, mbase = mtile * 16;
  v8f acc[4];
#pragma unroll
  for (int i = 0; i < 4; ++i) acc[i] = vzero();
  gemm_core4(A, lda, W, ldw, mbase, nbase, K, acc);
  int lane = threadIdx.x & 31, half = lane >> 4;
#pragma unroll
  for (int nt = 0; nt < 4; ++nt) {
    int col = nbase + nt * 16 + (lane & 15);
    float bb = bias ? bias[col] : 0.f;
#pragma unroll
    for (int r = 0; r < 8; ++r) {
      int row = mbase + r + half * 8;
      float v = acc[nt][r] + bb;
      if (res) v += res[(size_t)row * ldo + col];
      out[(size_t)row * ldo + col] = v;
    }
  }
}

// Dual GEMM (w1,w2) fused SwiGLU -> bf16 out with padded stride HIDP.
// Out-of-range weight rows are CLAMPED (loads always valid, no branch in loop);
// epilogue stores 0 for col >= HID, which also zeroes the ff pad columns.
__global__ void k_gemm_swiglu(const unsigned short* __restrict__ A,
                              const unsigned short* __restrict__ W1,
                              const unsigned short* __restrict__ W2,
                              unsigned short* __restrict__ out, int NT) {
  int widx = blockIdx.x * 4 + (threadIdx.x >> 5);
  int ntile = widx % NT, mtile = widx / NT;
  int nbase = ntile * 64, mbase = mtile * 16;
  int lane = threadIdx.x & 31, m = lane & 15, half = lane >> 4;
  v8f a1[4], a2[4];
#pragma unroll
  for (int i = 0; i < 4; ++i) { a1[i] = vzero(); a2[i] = vzero(); }
  const unsigned short* arow = A + (size_t)(mbase + m) * C_;
  const unsigned short* w1r[4];
  const unsigned short* w2r[4];
#pragma unroll
  for (int nt = 0; nt < 4; ++nt) {
    int n = nbase + nt * 16 + m;
    int nc = n < HID_ ? n : (HID_ - 1);               // clamp: always a valid row
    w1r[nt] = W1 + (size_t)nc * C_;
    w2r[nt] = W2 + (size_t)nc * C_;
  }
  for (int k = 0; k < C_; k += 32) {
    Frag a; loadA(a, arow, k, half);
#pragma unroll
    for (int nt = 0; nt < 4; ++nt) {
      Frag b1, b2;
      loadB(b1, w1r[nt], k, half);
      loadB(b2, w2r[nt], k, half);
      a1[nt] = wmma_bf(a, b1, a1[nt]);
      a2[nt] = wmma_bf(a, b2, a2[nt]);
    }
  }
#pragma unroll
  for (int nt = 0; nt < 4; ++nt) {
    int col = nbase + nt * 16 + m;
    if (col >= HIDP_) continue;
    bool valid = (col < HID_);
#pragma unroll
    for (int r = 0; r < 8; ++r) {
      int row = mbase + r + half * 8;
      float x1 = a1[nt][r], x2 = a2[nt][r];
      float sv = x1 / (1.f + __expf(-x1));            // silu
      float v = valid ? sv * x2 : 0.f;                // pad cols forced to 0
      out[(size_t)row * HIDP_ + col] = f2bf(v);
    }
  }
}

// ---------------- Flash attention: one wave per 16-row q-tile ----------------
__global__ void k_attn(const unsigned short* __restrict__ qp,
                       const unsigned short* __restrict__ kp,
                       const unsigned short* __restrict__ vT,
                       const float* __restrict__ meanrel,
                       unsigned short* __restrict__ ctxb) {
  __shared__ unsigned short Pt[4][16 * 32];           // per-wave P staging
  int wave = threadIdx.x >> 5;
  int lane = threadIdx.x & 31;
  int half = lane >> 4;
  int bh = blockIdx.x;                                // 0..B*H-1
  int qtile = blockIdx.y * 4 + wave;                  // 0..T/16-1
  int qbase = qtile * 16;
  const unsigned short* Qb = qp + (size_t)bh * T_ * HD_;
  const unsigned short* Kb = kp + (size_t)bh * T_ * HD_;
  const unsigned short* Vb = vT + (size_t)bh * HD_ * T_;

  Frag aQ0, aQ1;                                      // Q tile, K-dim = HD split 0..31 / 32..63
  const unsigned short* qrow = Qb + (size_t)(qbase + (lane & 15)) * HD_;
  loadA(aQ0, qrow, 0, half);
  loadA(aQ1, qrow, 32, half);

  v8f acc[4];
  float mrow[8], lrow[8];
#pragma unroll
  for (int i = 0; i < 4; ++i) acc[i] = vzero();
#pragma unroll
  for (int r = 0; r < 8; ++r) { mrow[r] = -1e30f; lrow[r] = 0.f; }

  unsigned short* Pm = Pt[wave];

  for (int kb = 0; kb < T_; kb += 32) {
#pragma unroll
    for (int sub = 0; sub < 2; ++sub) {
      int k0 = kb + sub * 16;
      Frag bK0, bK1;
      const unsigned short* krow = Kb + (size_t)(k0 + (lane & 15)) * HD_;
      loadB(bK0, krow, 0, half);
      loadB(bK1, krow, 32, half);
      v8f s = vzero();
      s = wmma_bf(aQ0, bK0, s);
      s = wmma_bf(aQ1, bK1, s);
      int kpos = k0 + (lane & 15);
#pragma unroll
      for (int r = 0; r < 8; ++r) {                   // relative-position bias
        int qr = qbase + r + half * 8;
        int idx = qr - kpos + (T_ - 1);
        idx = idx < 0 ? 0 : (idx > 2 * T_ - 2 ? 2 * T_ - 2 : idx);
        s[r] += meanrel[idx];
      }
#pragma unroll
      for (int r = 0; r < 8; ++r) {                   // online softmax (half-wave rows)
        float v = s[r];
        v = fmaxf(v, __shfl_xor(v, 1, 32));
        v = fmaxf(v, __shfl_xor(v, 2, 32));
        v = fmaxf(v, __shfl_xor(v, 4, 32));
        v = fmaxf(v, __shfl_xor(v, 8, 32));
        float mn = fmaxf(mrow[r], v);
        float sc = exp2f((mrow[r] - mn) * L2E_);
        float p  = exp2f((s[r] - mn) * L2E_);
        float ps = p;
        ps += __shfl_xor(ps, 1, 32); ps += __shfl_xor(ps, 2, 32);
        ps += __shfl_xor(ps, 4, 32); ps += __shfl_xor(ps, 8, 32);
        lrow[r] = lrow[r] * sc + ps;
        mrow[r] = mn;
#pragma unroll
        for (int nt = 0; nt < 4; ++nt) acc[nt][r] *= sc;
        Pm[(r + half * 8) * 32 + sub * 16 + (lane & 15)] = f2bf(p);
      }
    }
    asm volatile("s_wait_dscnt 0" ::: "memory");      // DS stores visible before transposed reload
    Frag aP;
    {
      const unsigned short* pr = Pm + (lane & 15) * 32;
      int koff = half * 8;
      aP.u[0] = *reinterpret_cast<const uint4*>(pr + koff);
      aP.u[1] = *reinterpret_cast<const uint4*>(pr + koff + 16);
    }
#pragma unroll
    for (int nt = 0; nt < 4; ++nt) {
      Frag bV;
      const unsigned short* vrow = Vb + (size_t)(nt * 16 + (lane & 15)) * T_;
      loadB(bV, vrow, kb, half);
      acc[nt] = wmma_bf(aP, bV, acc[nt]);
    }
    asm volatile("s_wait_dscnt 0" ::: "memory");      // protect WAR for next iteration
  }

  int h = bh & (H_ - 1), b = bh >> 4;
#pragma unroll
  for (int r = 0; r < 8; ++r) {
    float inv = 1.f / lrow[r];
    int row = qbase + r + half * 8;
    size_t base = ((size_t)b * T_ + row) * C_ + h * HD_;
#pragma unroll
    for (int nt = 0; nt < 4; ++nt)
      ctxb[base + nt * 16 + (lane & 15)] = f2bf(acc[nt][r] * inv);
  }
}

// ---------------- launch ----------------
extern "C" void kernel_launch(void* const* d_in, const int* in_sizes, int n_in,
                              void* d_out, int out_size, void* d_ws, size_t ws_size,
                              hipStream_t stream) {
  const float* query     = (const float*)d_in[0];
  const float* key       = (const float*)d_in[1];
  const float* g_q       = (const float*)d_in[2];
  const float* b_q       = (const float*)d_in[3];
  const float* g_k       = (const float*)d_in[4];
  const float* b_k       = (const float*)d_in[5];
  const float* in_proj_w = (const float*)d_in[6];
  const float* in_proj_b = (const float*)d_in[7];
  const float* out_w     = (const float*)d_in[8];
  const float* out_b     = (const float*)d_in[9];
  const float* proj_w    = (const float*)d_in[10];
  const float* proj_b    = (const float*)d_in[11];
  const float* rel_table = (const float*)d_in[12];
  const float* g_ffn     = (const float*)d_in[13];
  const float* b_ffn     = (const float*)d_in[14];
  const float* w1        = (const float*)d_in[15];
  const float* w2        = (const float*)d_in[16];
  const float* w3        = (const float*)d_in[17];
  const float* g_out     = (const float*)d_in[18];
  const float* b_out     = (const float*)d_in[19];
  float* out = (float*)d_out;

  char* p = (char*)d_ws;
  auto alloc = [&](size_t bytes) -> void* {
    void* r = p; p += (bytes + 255) & ~(size_t)255; return r;
  };
  unsigned short* qln    = (unsigned short*)alloc((size_t)BT_ * C_ * 2);
  unsigned short* kln    = (unsigned short*)alloc((size_t)BT_ * C_ * 2);
  unsigned short* wqkvb  = (unsigned short*)alloc((size_t)3 * C_ * C_ * 2);
  unsigned short* outwb  = (unsigned short*)alloc((size_t)C_ * C_ * 2);
  unsigned short* projwb = (unsigned short*)alloc((size_t)C_ * C_ * 2);
  unsigned short* w1b    = (unsigned short*)alloc((size_t)HID_ * C_ * 2);
  unsigned short* w2b    = (unsigned short*)alloc((size_t)HID_ * C_ * 2);
  unsigned short* w3b    = (unsigned short*)alloc((size_t)C_ * HIDP_ * 2);
  unsigned short* qp     = (unsigned short*)alloc((size_t)BT_ * C_ * 2);
  unsigned short* kp     = (unsigned short*)alloc((size_t)BT_ * C_ * 2);
  unsigned short* vT     = (unsigned short*)alloc((size_t)BT_ * C_ * 2);
  unsigned short* ctxb   = (unsigned short*)alloc((size_t)BT_ * C_ * 2);
  unsigned short* attn1  = (unsigned short*)alloc((size_t)BT_ * C_ * 2);
  unsigned short* hln    = (unsigned short*)alloc((size_t)BT_ * C_ * 2);
  unsigned short* ff     = (unsigned short*)alloc((size_t)BT_ * HIDP_ * 2);
  float*          xbuf   = (float*)alloc((size_t)BT_ * C_ * 4);
  float*          mrel   = (float*)alloc((size_t)(2 * T_ - 1) * 4);

  // 1) LayerNorm q,k -> bf16
  k_ln<1><<<BT_, 256, 0, stream>>>(query, g_q, b_q, qln);
  k_ln<1><<<BT_, 256, 0, stream>>>(key,   g_k, b_k, kln);
  // 2) weight converts
  k_conv<<<(3 * C_ * C_ + 255) / 256, 256, 0, stream>>>(in_proj_w, wqkvb, 3 * C_ * C_);
  k_conv<<<(C_ * C_ + 255) / 256, 256, 0, stream>>>(out_w,  outwb,  C_ * C_);
  k_conv<<<(C_ * C_ + 255) / 256, 256, 0, stream>>>(proj_w, projwb, C_ * C_);
  k_conv<<<(HID_ * C_ + 255) / 256, 256, 0, stream>>>(w1, w1b, HID_ * C_);
  k_conv<<<(HID_ * C_ + 255) / 256, 256, 0, stream>>>(w2, w2b, HID_ * C_);
  k_convpad<<<(C_ * HIDP_ + 255) / 256, 256, 0, stream>>>(w3, w3b);
  k_meanrel<<<(2 * T_ - 1 + 255) / 256, 256, 0, stream>>>(rel_table, mrel);
  // 3) QKV projection (q scaled by 1/sqrt(HD); v written transposed)
  { int NT = 3 * C_ / 64, waves = (BT_ / 16) * NT;
    k_gemm_qkv<<<waves / 4, 128, 0, stream>>>(qln, kln, wqkvb, in_proj_b, qp, kp, vT); }
  // 4) flash attention
  { dim3 g(B_ * H_, T_ / 16 / 4);
    k_attn<<<g, 128, 0, stream>>>(qp, kp, vT, mrel, ctxb); }
  // 5) ctx @ out_w.T + out_b -> bf16
  { int NT = C_ / 64, waves = (BT_ / 16) * NT;
    k_gemm_bf16out<<<waves / 4, 128, 0, stream>>>(ctxb, C_, outwb, C_, C_,
                                                  out_b, attn1, C_, NT); }
  // 6) attn1 @ proj_w.T + proj_b + query -> xbuf (f32)
  { int NT = C_ / 64, waves = (BT_ / 16) * NT;
    k_gemm_f32res<<<waves / 4, 128, 0, stream>>>(attn1, C_, projwb, C_, C_,
                                                 proj_b, query, xbuf, C_, NT); }
  // 7) LN -> hln bf16
  k_ln<1><<<BT_, 256, 0, stream>>>(xbuf, g_ffn, b_ffn, hln);
  // 8) fused SwiGLU: silu(h@w1.T) * (h@w2.T) -> ff bf16 (padded)
  { int NT = (HID_ + 63) / 64, waves = (BT_ / 16) * NT;
    k_gemm_swiglu<<<waves / 4, 128, 0, stream>>>(hln, w1b, w2b, ff, NT); }
  // 9) ff @ w3.T + xbuf -> xbuf (in place, per-element read-then-write)
  { int NT = C_ / 64, waves = (BT_ / 16) * NT;
    k_gemm_f32res<<<waves / 4, 128, 0, stream>>>(ff, HIDP_, w3b, HIDP_, C_,
                                                 nullptr, xbuf, xbuf, C_, NT); }
  // 10) final LN -> d_out (f32)
  k_ln<0><<<BT_, 256, 0, stream>>>(xbuf, g_out, b_out, out);
}
